// Contraction_47004122087964
// MI455X (gfx1250) — compile-verified, matrix operand order checked by reading, not run on hardware
//
#include <hip/hip_runtime.h>

typedef _Float16 f16;
typedef __attribute__((ext_vector_type(16))) _Float16 v16h;
typedef __attribute__((ext_vector_type(8)))  _Float16 v8h;
typedef __attribute__((ext_vector_type(2)))  _Float16 v2h;
typedef __attribute__((ext_vector_type(8)))  float    v8f;

#define NNODES 3000
#define CFEAT  128
#define YD     16
#define ED     10
#define Z3D    23
#define Z2D    4
#define KU3    368            // 16*23  (i,kz) pairs
#define KTOT   384            // + 4 U2 columns, padded to 12*32
#define KSTEPS 12
#define NTILES 16             // 256 xy columns / 16
#define FRAG_HALVES (KSTEPS*NTILES*32*16)   // 98304 f16 = 192 KB

// ---------------------------------------------------------------------------
// Pack U3 (+U2 tail) into per-lane WMMA B-fragment order:
//   half index = ((ks*16 + nt)*32 + lane)*16 + (dword*2 + half)
//   B[kflat][xy]: kflat = ks*32 + klocal, xy = nt*16 + (lane&15), x=nt, y=lane&15
//   klocal = (d<4?0:16) + (d&3)*2 + h + (lane>>4)*8   (mirrors 16-bit A layout)
// ---------------------------------------------------------------------------
__global__ void pack_B(const float* __restrict__ U3,
                       const float* __restrict__ U2,
                       f16* __restrict__ Bp) {
  int idx = blockIdx.x * blockDim.x + threadIdx.x;
  if (idx >= FRAG_HALVES) return;
  int ks   = idx >> 13;        // /(16*32*16)
  int rem  = idx & 8191;
  int nt   = rem >> 9;         // /(32*16)
  int rem2 = rem & 511;
  int lane = rem2 >> 4;
  int pos  = rem2 & 15;
  int d = pos >> 1, h = pos & 1;
  int g = lane >> 4, n = lane & 15;
  int klocal = ((d < 4) ? 0 : 16) + (d & 3) * 2 + h + g * 8;
  int kflat  = ks * 32 + klocal;
  int x = nt, y = n;
  float v = 0.0f;
  if (kflat < KU3) {
    int i  = kflat / Z3D;
    int kz = kflat - i * Z3D;
    v = U3[((x * 16 + y) * 16 + i) * Z3D + kz];          // U3[0][x][y][i][kz]
  } else if (kflat < KU3 + Z2D) {
    v = U2[(x * 16 + y) * Z2D + (kflat - KU3)];          // U2[0][x][y][k2]
  }
  Bp[idx] = (f16)v;
}

// ---------------------------------------------------------------------------
// One workgroup per node. 8 waves; wave w owns xy-tiles {2w, 2w+1} with B
// fragments resident in VGPRs. Inner GEMM: (16 c) x (K=384) x (256 xy) in
// v_wmma_f32_16x16x32_f16 with double-buffered A fragments; epilogue applies
// emb_x*emb_y weights + cross-lane reduce.
// ---------------------------------------------------------------------------
__global__ void __launch_bounds__(256) mace_node_kernel(
    const float* __restrict__ emb,    // (N,128,16)
    const float* __restrict__ attr,   // (N,10)
    const float* __restrict__ U1,     // (1,16,1)
    const float* __restrict__ W3,     // (10,23,128)
    const float* __restrict__ W2,     // (10,4,128)
    const float* __restrict__ W1,     // (10,1,128)
    const f16*   __restrict__ Bp,     // packed B fragments
    float* __restrict__ out)          // (N,128)
{
  __shared__ float emb_s[CFEAT * YD];            // 8 KB
  __shared__ float w3n_s[Z3D * CFEAT];           // 11.5 KB
  __shared__ float w2n_s[Z2D * CFEAT];           // 2 KB
  __shared__ float w1n_s[CFEAT];
  __shared__ float u1_s[YD];
  __shared__ float acc_s[CFEAT];
  __shared__ __align__(16) f16 Afull[16 * KTOT]; // 12 KB

  const int b    = blockIdx.x;
  const int t    = threadIdx.x;
  const int lane = t & 31;
  const int wv   = t >> 5;                       // 0..7

  // --- B fragments: resident in registers for the whole kernel ------------
  v16h Bfrag[2][KSTEPS];
  const v16h* Bv = (const v16h*)Bp;
  #pragma unroll
  for (int tt = 0; tt < 2; ++tt) {
    const int nt = wv * 2 + tt;
    #pragma unroll
    for (int ks = 0; ks < KSTEPS; ++ks)
      Bfrag[tt][ks] = Bv[(ks * NTILES + nt) * 32 + lane];
  }

  // --- per-node weights: w* = attr . W*  (attr uniform -> scalar loads) ---
  float at[ED];
  #pragma unroll
  for (int e = 0; e < ED; ++e) at[e] = attr[b * ED + e];

  for (int j = t; j < CFEAT * YD; j += 256) emb_s[j] = emb[b * CFEAT * YD + j];
  for (int j = t; j < Z3D * CFEAT; j += 256) {
    int kz = j >> 7, c = j & 127;
    float s = 0.0f;
    #pragma unroll
    for (int e = 0; e < ED; ++e) s += at[e] * W3[(e * Z3D + kz) * CFEAT + c];
    w3n_s[j] = s;
  }
  for (int j = t; j < Z2D * CFEAT; j += 256) {
    int k = j >> 7, c = j & 127;
    float s = 0.0f;
    #pragma unroll
    for (int e = 0; e < ED; ++e) s += at[e] * W2[(e * Z2D + k) * CFEAT + c];
    w2n_s[j] = s;
  }
  if (t < CFEAT) {
    float s = 0.0f;
    #pragma unroll
    for (int e = 0; e < ED; ++e) s += at[e] * W1[e * CFEAT + t];
    w1n_s[t] = s;
    acc_s[t] = 0.0f;
  }
  if (t < YD) u1_s[t] = U1[t];

  const int m  = lane & 15;      // A-matrix row (c within tile)
  const int g  = lane >> 4;      // K-group (A) / M-group (C/D)
  const int yy = lane & 15;      // N column of C/D = y index
  const int x0 = wv * 2, x1 = wv * 2 + 1;

  // A-panel element value at flattened K index k for feature channel c
  auto a_elem = [&](int c, int k) -> float {
    if (k < KU3) {
      int i  = k / Z3D;
      int kz = k - i * Z3D;
      return emb_s[c * YD + i] * w3n_s[kz * CFEAT + c];
    } else if (k < KU3 + Z2D) {
      return w2n_s[(k - KU3) * CFEAT + c];
    }
    return 0.0f;
  };

  // A fragment for K-step ks (halves 0-7 = K[ks*32+g*8..], 8-15 = K[+16])
  auto load_a = [&](int ks) -> v16h {
    v8h lo = *(const v8h*)&Afull[m * KTOT + ks * 32 + g * 8];
    v8h hi = *(const v8h*)&Afull[m * KTOT + ks * 32 + 16 + g * 8];
    v16h a;
    #pragma unroll
    for (int j = 0; j < 8; ++j) { a[j] = lo[j]; a[j + 8] = hi[j]; }
    return a;
  };

  for (int mt = 0; mt < 8; ++mt) {
    const int c0 = mt * 16;
    __syncthreads();             // prior Afull fully consumed / LDS init done

    // build A panel (paired f16 stores): A[c, i*23+kz] = emb[c,i]*w3[kz,c];
    // A[c, 368+k] = w2[k,c]; zero padding to 384.
    for (int j = t; j < 16 * (KTOT / 2); j += 256) {
      int mm = j / (KTOT / 2);
      int k  = (j - mm * (KTOT / 2)) * 2;
      int c  = c0 + mm;
      v2h p;
      p[0] = (f16)a_elem(c, k);
      p[1] = (f16)a_elem(c, k + 1);
      *(v2h*)&Afull[mm * KTOT + k] = p;
    }
    __syncthreads();

    v8f acc0 = {};
    v8f acc1 = {};
    v16h acur = load_a(0);
    #pragma unroll
    for (int ks = 0; ks < KSTEPS; ++ks) {
      v16h anext;
      if (ks + 1 < KSTEPS) anext = load_a(ks + 1);   // prefetch next K-step
      acc0 = __builtin_amdgcn_wmma_f32_16x16x32_f16(
          false, acur, false, Bfrag[0][ks], (short)0, acc0, false, false);
      acc1 = __builtin_amdgcn_wmma_f32_16x16x32_f16(
          false, acur, false, Bfrag[1][ks], (short)0, acc1, false, false);
      if (ks + 1 < KSTEPS) acur = anext;
    }

    // epilogue: acc_s[c] += sum_{x,y} acc[c,xy] * emb[c,x] * emb[c,y]
    #pragma unroll
    for (int r = 0; r < 8; ++r) {
      int c = c0 + r + 8 * g;
      float ey  = emb_s[c * YD + yy];
      float ex0 = emb_s[c * YD + x0];
      float ex1 = emb_s[c * YD + x1];
      float s = ey * (acc0[r] * ex0 + acc1[r] * ex1);
      #pragma unroll
      for (int off = 1; off < 16; off <<= 1)
        s += __shfl_xor(s, off, 16);
      if (yy == 0) atomicAdd(&acc_s[c], s);
    }
  }

  __syncthreads();
  if (t < CFEAT) {
    float u1dot = 0.0f;
    #pragma unroll
    for (int xx = 0; xx < YD; ++xx) u1dot += u1_s[xx] * emb_s[t * YD + xx];
    out[b * CFEAT + t] = acc_s[t] + w1n_s[t] * u1dot;
  }
}

extern "C" void kernel_launch(void* const* d_in, const int* in_sizes, int n_in,
                              void* d_out, int out_size, void* d_ws, size_t ws_size,
                              hipStream_t stream) {
  (void)in_sizes; (void)n_in; (void)out_size; (void)ws_size;
  const float* emb  = (const float*)d_in[0];
  const float* attr = (const float*)d_in[1];
  const float* U3   = (const float*)d_in[2];
  const float* U2   = (const float*)d_in[3];
  const float* U1   = (const float*)d_in[4];
  const float* W3   = (const float*)d_in[5];
  const float* W2   = (const float*)d_in[6];
  const float* W1   = (const float*)d_in[7];
  f16* Bp = (f16*)d_ws;   // needs 192 KB of workspace

  pack_B<<<(FRAG_HALVES + 255) / 256, 256, 0, stream>>>(U3, U2, Bp);
  mace_node_kernel<<<NNODES, 256, 0, stream>>>(emb, attr, U1, W3, W2, W1, Bp,
                                               (float*)d_out);
}
